// DBTransformerLayer_36069135352083
// MI455X (gfx1250) — matrix operand order repaired and checked by reading, not compiled, not run on hardware
//
#include <hip/hip_runtime.h>
#include <hip/hip_bf16.h>

typedef __attribute__((ext_vector_type(16))) _Float16 v16h;
typedef __attribute__((ext_vector_type(8)))  _Float16 v8h;
typedef __attribute__((ext_vector_type(8)))  float    v8f;

#define NA_  20000
#define NB_  20000
#define T_   4
#define D_   128
#define FF_  64
#define E_   100000
#define NTOK (T_ * D_)   // 512 values per node

// ---------------- WMMA helpers (CDNA5 16x16x32 f16, f32 accum) ----------------

__device__ __forceinline__ v8f zero8() {
  v8f z;
#pragma unroll
  for (int i = 0; i < 8; ++i) z[i] = 0.f;
  return z;
}

__device__ __forceinline__ v8f wmma_f16(v16h a, v16h b, v8f c) {
  // (neg_a, A, neg_b, B, c_mod, C, reuse_a, reuse_b)
  return __builtin_amdgcn_wmma_f32_16x16x32_f16(false, a, false, b, (short)0, c,
                                                false, false);
}

// A fragment: 16x32 f16 tile (rows = lane&15), contiguous k order: element s holds k = 16*(lane>>4) + s.
__device__ __forceinline__ v16h load_a(const _Float16* base, int stride, int kbase,
                                       int lane) {
  int row = lane & 15;
  int off = lane & 16;  // 0 or 16
  const _Float16* p = base + row * stride + kbase + off;
  v8h lo = *(const v8h*)p;
  v8h hi = *(const v8h*)(p + 8);
  v16h r;
#pragma unroll
  for (int i = 0; i < 8; ++i) { r[i] = lo[i]; r[i + 8] = hi[i]; }
  return r;
}

// B fragment: 32x16 tile of W^T, i.e. B[k][n] = W[nbase+n][kbase+k].
// Lanes 0..15: K=0..15 of column n=lane; lanes 16..31: K=16..31.
__device__ __forceinline__ v16h load_b(const _Float16* W, int ldk, int nbase,
                                       int kbase, int lane) {
  int n = nbase + (lane & 15);
  int k = kbase + (lane & 16);
  return *(const v16h*)(W + (size_t)n * ldk + k);
}

// Generic row-tile GEMM: D[16 x 16*NT] = A[16 x 32*KT] @ W^T, with B-fragment
// double buffering so the next tile's global loads overlap the WMMA chain.
template <int NT, int KT, typename EpiFn>
__device__ __forceinline__ void gemm_rows(const v16h* a, const _Float16* __restrict__ W,
                                          int ldk, int lane, EpiFn epi) {
  v16h bcur[KT], bnext[KT];
#pragma unroll
  for (int kt = 0; kt < KT; ++kt) bcur[kt] = load_b(W, ldk, 0, kt * 32, lane);
#pragma unroll 4
  for (int nt = 0; nt < NT; ++nt) {
#pragma unroll
    for (int kt = 0; kt < KT; ++kt)
      bnext[kt] = (nt + 1 < NT) ? load_b(W, ldk, (nt + 1) * 16, kt * 32, lane)
                                : bcur[kt];
    v8f acc = zero8();
#pragma unroll
    for (int kt = 0; kt < KT; ++kt) acc = wmma_f16(a[kt], bcur[kt], acc);
    epi(nt, acc);
#pragma unroll
    for (int kt = 0; kt < KT; ++kt) bcur[kt] = bnext[kt];
  }
}

// ---------------- small utility kernels ----------------

__global__ void cvt_f16_kernel(const float* __restrict__ in,
                               _Float16* __restrict__ out, int n) {
  int i = blockIdx.x * 256 + threadIdx.x;
  if (i < n) out[i] = (_Float16)in[i];
}

__global__ void zero_f32_kernel(float* __restrict__ p, int n) {
  int i = blockIdx.x * 256 + threadIdx.x;
  if (i < n) p[i] = 0.f;
}

__global__ void finalize_kernel(float* __restrict__ out,
                                const float* __restrict__ cnt, int n) {
  int i = blockIdx.x * 256 + threadIdx.x;
  if (i < n) out[i] /= fmaxf(cnt[i / NTOK], 1.f);
}

// ---------------- main relation kernel ----------------
// Block = 64 threads = 2 waves; each wave processes 2 edges (16 sequence rows).
// Rows per wave: e*8 + t, t=0..3 dst tokens (x_i), t=4..7 projected src (x_j).
// LDS: 8KB sX + 8KB sT + 24KB pool (sQKV aliased with sF) = 40KB/block
//  -> 8 blocks per WGP (320KB LDS), 16 waves, 4 waves/SIMD.

__global__ __launch_bounds__(64)
void rel_kernel(const _Float16* __restrict__ xs,   // src nodes f16 [Ns,T,D]
                const _Float16* __restrict__ xd,   // dst nodes f16 [Nd,T,D]
                const int* __restrict__ edge,      // [2,E]: row0=src, row1=dst
                const _Float16* __restrict__ Wb,   // [D,D]
                const _Float16* __restrict__ Wi,   // [3D,D]
                const _Float16* __restrict__ Wo,   // [D,D]
                const _Float16* __restrict__ W1,   // [FF,D]
                const _Float16* __restrict__ W2,   // [D,FF]
                const float* __restrict__ bb,  const float* __restrict__ bi,
                const float* __restrict__ bo,  const float* __restrict__ b1,
                const float* __restrict__ b2,
                const float* __restrict__ n1w, const float* __restrict__ n1b,
                const float* __restrict__ n2w, const float* __restrict__ n2b,
                float* __restrict__ outp, float* __restrict__ cnt, int nE)
{
  __shared__ _Float16 sX[2][16][D_];  // x_c / x1 (post-LN1)
  __shared__ _Float16 sT[2][16][D_];  // src staging / attn out / ff hidden
  // sQKV (16x384 f16 = 12KB/wave) is dead before sF (16x128 f32 = 8KB/wave)
  // is first written -> alias them in one pool.
  __shared__ __align__(16) unsigned char sPool[2][16 * 3 * D_ * sizeof(_Float16)];

  const int wave = threadIdx.x >> 5;
  const int lane = threadIdx.x & 31;

  _Float16 (*mX)[D_]     = sX[wave];
  _Float16 (*mT)[D_]     = sT[wave];
  _Float16 (*mQ)[3 * D_] = (_Float16 (*)[3 * D_])sPool[wave];
  float    (*mF)[D_]     = (float (*)[D_])sPool[wave];

  long eb = (long)blockIdx.x * 4 + wave * 2;
  bool v0 = (eb < nE), v1 = (eb + 1 < nE);
  int e0 = (int)(v0 ? eb : (nE - 1));
  int e1 = (int)(v1 ? eb + 1 : (nE - 1));
  int src0 = edge[e0],      src1 = edge[e1];
  int dst0 = edge[nE + e0], dst1 = edge[nE + e1];

  // ---- stage: src tokens -> mT rows 0..7 (rows 8..15 zeroed);
  //             dst tokens -> mX rows {0..3, 8..11}
  {
    int c = lane * 4;  // 4 contiguous f16 per lane per row
#pragma unroll
    for (int j = 0; j < 8; ++j) {
      int s   = (j < 4) ? src0 : src1;
      int d   = (j < 4) ? dst0 : dst1;
      int tok = j & 3;
      const _Float16* ps = xs + ((size_t)s * T_ + tok) * D_;
      const _Float16* pd = xd + ((size_t)d * T_ + tok) * D_;
      int xrow = (j >> 2) * 8 + tok;
#pragma unroll
      for (int u = 0; u < 4; ++u) {
        mT[j][c + u]     = ps[c + u];
        mT[8 + j][c + u] = (_Float16)0.f;
        mX[xrow][c + u]  = pd[c + u];
      }
    }
  }
  __syncthreads();

  const int N     = lane & 15;
  const int mbase = (lane >> 4) * 8;
  v16h a[4];

  // ---- b_proj: src rows @ Wb^T + bb -> mX rows {4..7, 12..15}
#pragma unroll
  for (int kt = 0; kt < 4; ++kt) a[kt] = load_a(&mT[0][0], D_, kt * 32, lane);
  gemm_rows<8, 4>(a, Wb, D_, lane, [&](int nt, v8f acc) {
    float bias = bb[nt * 16 + N];
    if (mbase == 0) {  // only rows 0..7 are valid src rows
#pragma unroll
      for (int r = 0; r < 8; ++r) {
        int xrow = (r >> 2) * 8 + 4 + (r & 3);
        mX[xrow][nt * 16 + N] = (_Float16)(acc[r] + bias);
      }
    }
  });
  __syncthreads();

  // ---- qkv projection: mX(16x128) @ Wi^T(128x384) + bi -> mQ
#pragma unroll
  for (int kt = 0; kt < 4; ++kt) a[kt] = load_a(&mX[0][0], D_, kt * 32, lane);
  gemm_rows<24, 4>(a, Wi, D_, lane, [&](int nt, v8f acc) {
    float bias = bi[nt * 16 + N];
#pragma unroll
    for (int r = 0; r < 8; ++r)
      mQ[mbase + r][nt * 16 + N] = (_Float16)(acc[r] + bias);
  });
  __syncthreads();

  // ---- attention (tiny: 8x8 per head, VALU f32). 2 edges * 8 heads * 8 q = 128 tasks.
  const float scale = 0.25f;  // 1/sqrt(DH=16)
#pragma unroll
  for (int it = 0; it < 4; ++it) {
    int task = it * 32 + lane;
    int e = task >> 6, h = (task >> 3) & 7, q = task & 7;
    int row = e * 8 + q;
    float qv[16];
#pragma unroll
    for (int d = 0; d < 16; ++d) qv[d] = (float)mQ[row][h * 16 + d];
    float s[8];
    float m = -1e30f;
#pragma unroll
    for (int k = 0; k < 8; ++k) {
      float acc = 0.f;
#pragma unroll
      for (int d = 0; d < 16; ++d)
        acc += qv[d] * (float)mQ[e * 8 + k][D_ + h * 16 + d];
      s[k] = acc * scale;
      m = fmaxf(m, s[k]);
    }
    float sum = 0.f;
#pragma unroll
    for (int k = 0; k < 8; ++k) { s[k] = __expf(s[k] - m); sum += s[k]; }
    float inv = 1.f / sum;
    float o[16];
#pragma unroll
    for (int d = 0; d < 16; ++d) o[d] = 0.f;
#pragma unroll
    for (int k = 0; k < 8; ++k) {
      float w = s[k] * inv;
#pragma unroll
      for (int d = 0; d < 16; ++d)
        o[d] += w * (float)mQ[e * 8 + k][2 * D_ + h * 16 + d];
    }
#pragma unroll
    for (int d = 0; d < 16; ++d) mT[row][h * 16 + d] = (_Float16)o[d];
  }
  __syncthreads();

  // ---- out_proj + residual -> mF (f32; mQ is dead from here on)
#pragma unroll
  for (int kt = 0; kt < 4; ++kt) a[kt] = load_a(&mT[0][0], D_, kt * 32, lane);
  gemm_rows<8, 4>(a, Wo, D_, lane, [&](int nt, v8f acc) {
    float bias = bo[nt * 16 + N];
#pragma unroll
    for (int r = 0; r < 8; ++r) {
      int M = mbase + r, col = nt * 16 + N;
      mF[M][col] = acc[r] + bias + (float)mX[M][col];
    }
  });
  __syncthreads();

  // ---- LayerNorm 1 -> mX (f16). Lane pair (lane, lane^16) shares a row.
  {
    int row = lane & 15;
    int cb  = (lane >> 4) * 64;
    float sum = 0.f, sq = 0.f;
    for (int c2 = 0; c2 < 64; ++c2) {
      float v = mF[row][cb + c2];
      sum += v; sq += v * v;
    }
    sum += __shfl_xor(sum, 16);
    sq  += __shfl_xor(sq, 16);
    float mean = sum * (1.f / 128.f);
    float var  = sq * (1.f / 128.f) - mean * mean;
    float rstd = rsqrtf(var + 1e-5f);
    for (int c2 = 0; c2 < 64; ++c2) {
      int col = cb + c2;
      mX[row][col] = (_Float16)((mF[row][col] - mean) * rstd * n1w[col] + n1b[col]);
    }
  }
  __syncthreads();

  // ---- FF1: relu(x1 @ W1^T + b1) -> mT (cols 0..63)
#pragma unroll
  for (int kt = 0; kt < 4; ++kt) a[kt] = load_a(&mX[0][0], D_, kt * 32, lane);
  gemm_rows<4, 4>(a, W1, D_, lane, [&](int nt, v8f acc) {
    float bias = b1[nt * 16 + N];
#pragma unroll
    for (int r = 0; r < 8; ++r)
      mT[mbase + r][nt * 16 + N] = (_Float16)fmaxf(acc[r] + bias, 0.f);
  });
  __syncthreads();

  // ---- FF2 + residual -> mF (f32)
  {
    v16h a2[2];
#pragma unroll
    for (int kt = 0; kt < 2; ++kt) a2[kt] = load_a(&mT[0][0], D_, kt * 32, lane);
    gemm_rows<8, 2>(a2, W2, FF_, lane, [&](int nt, v8f acc) {
      float bias = b2[nt * 16 + N];
#pragma unroll
      for (int r = 0; r < 8; ++r) {
        int M = mbase + r, col = nt * 16 + N;
        mF[M][col] = acc[r] + bias + (float)mX[M][col];
      }
    });
  }
  __syncthreads();

  // ---- LayerNorm 2 fused with scatter-add of the first T_ tokens per edge
  {
    int row = lane & 15;
    int cb  = (lane >> 4) * 64;
    float sum = 0.f, sq = 0.f;
    for (int c2 = 0; c2 < 64; ++c2) {
      float v = mF[row][cb + c2];
      sum += v; sq += v * v;
    }
    sum += __shfl_xor(sum, 16);
    sq  += __shfl_xor(sq, 16);
    float mean = sum * (1.f / 128.f);
    float var  = sq * (1.f / 128.f) - mean * mean;
    float rstd = rsqrtf(var + 1e-5f);

    int e   = row >> 3;
    int tok = row & 7;
    bool val = e ? v1 : v0;
    int dst  = e ? dst1 : dst0;
    if (val && tok < T_) {
      float* op = outp + ((size_t)dst * T_ + tok) * D_;
      for (int c2 = 0; c2 < 64; ++c2) {
        int col = cb + c2;
        float v = (mF[row][col] - mean) * rstd * n2w[col] + n2b[col];
        atomicAdd(op + col, v);
      }
    }
    if (lane == 0 && v0) atomicAdd(cnt + dst0, 1.f);
    if (lane == 8 && v1) atomicAdd(cnt + dst1, 1.f);
  }
}

// ---------------- host launch ----------------

extern "C" void kernel_launch(void* const* d_in, const int* in_sizes, int n_in,
                              void* d_out, int out_size, void* d_ws, size_t ws_size,
                              hipStream_t stream) {
  (void)in_sizes; (void)n_in; (void)ws_size;

  const float* xA  = (const float*)d_in[0];
  const float* xB  = (const float*)d_in[1];
  const int*   eAB = (const int*)d_in[2];
  const int*   eBA = (const int*)d_in[3];
  const float* bw  = (const float*)d_in[4];
  const float* bb  = (const float*)d_in[5];
  const float* wi  = (const float*)d_in[6];
  const float* bi  = (const float*)d_in[7];
  const float* wo  = (const float*)d_in[8];
  const float* bo  = (const float*)d_in[9];
  const float* l1w = (const float*)d_in[10];
  const float* l1b = (const float*)d_in[11];
  const float* l2w = (const float*)d_in[12];
  const float* l2b = (const float*)d_in[13];
  const float* n1w = (const float*)d_in[14];
  const float* n1b = (const float*)d_in[15];
  const float* n2w = (const float*)d_in[16];
  const float* n2b = (const float*)d_in[17];
  float* out = (float*)d_out;

  // workspace layout (f16 caches + scatter counters), ~41.6 MB
  char* ws = (char*)d_ws;
  size_t off = 0;
  _Float16* xA16 = (_Float16*)(ws + off); off += (size_t)NA_ * NTOK * 2;
  _Float16* xB16 = (_Float16*)(ws + off); off += (size_t)NB_ * NTOK * 2;
  _Float16* wb16 = (_Float16*)(ws + off); off += 2ull * D_ * D_ * 2;
  _Float16* wi16 = (_Float16*)(ws + off); off += 2ull * 3 * D_ * D_ * 2;
  _Float16* wo16 = (_Float16*)(ws + off); off += 2ull * D_ * D_ * 2;
  _Float16* w116 = (_Float16*)(ws + off); off += 2ull * FF_ * D_ * 2;
  _Float16* w216 = (_Float16*)(ws + off); off += 2ull * D_ * FF_ * 2;
  float* cntA = (float*)(ws + off);
  float* cntB = cntA + NA_;

  cvt_f16_kernel<<<(NA_ * NTOK + 255) / 256, 256, 0, stream>>>(xA, xA16, NA_ * NTOK);
  cvt_f16_kernel<<<(NB_ * NTOK + 255) / 256, 256, 0, stream>>>(xB, xB16, NB_ * NTOK);
  cvt_f16_kernel<<<(2 * D_ * D_ + 255) / 256, 256, 0, stream>>>(bw, wb16, 2 * D_ * D_);
  cvt_f16_kernel<<<(2 * 3 * D_ * D_ + 255) / 256, 256, 0, stream>>>(wi, wi16, 2 * 3 * D_ * D_);
  cvt_f16_kernel<<<(2 * D_ * D_ + 255) / 256, 256, 0, stream>>>(wo, wo16, 2 * D_ * D_);
  cvt_f16_kernel<<<(2 * FF_ * D_ + 255) / 256, 256, 0, stream>>>(l1w, w116, 2 * FF_ * D_);
  cvt_f16_kernel<<<(2 * D_ * FF_ + 255) / 256, 256, 0, stream>>>(l2w, w216, 2 * D_ * FF_);

  zero_f32_kernel<<<(out_size + 255) / 256, 256, 0, stream>>>(out, out_size);
  zero_f32_kernel<<<(NA_ + NB_ + 255) / 256, 256, 0, stream>>>(cntA, NA_ + NB_);

  int blocks = (E_ + 3) / 4;  // 4 edges per 64-thread block (2 per wave)

  // relation 0: A -> B (updates out_B = out + NA*NTOK)
  rel_kernel<<<blocks, 64, 0, stream>>>(
      xA16, xB16, eAB, wb16, wi16, wo16, w116, w216,
      bb, bi, bo, l1b, l2b, n1w, n1b, n2w, n2b,
      out + (size_t)NA_ * NTOK, cntB, E_);

  // relation 1: B -> A (updates out_A = out)
  rel_kernel<<<blocks, 64, 0, stream>>>(
      xB16, xA16, eBA,
      wb16 + D_ * D_, wi16 + 3 * D_ * D_, wo16 + D_ * D_,
      w116 + FF_ * D_, w216 + D_ * FF_,
      bb + D_, bi + 3 * D_, bo + D_, l1b + FF_, l2b + D_,
      n1w + D_, n1b + D_, n2w + D_, n2b + D_,
      out, cntA, E_);

  finalize_kernel<<<(NA_ * NTOK + 255) / 256, 256, 0, stream>>>(out, cntA, NA_ * NTOK);
  finalize_kernel<<<(NB_ * NTOK + 255) / 256, 256, 0, stream>>>(
      out + (size_t)NA_ * NTOK, cntB, NB_ * NTOK);
}
